// CrossAttention_37220186587177
// MI455X (gfx1250) — matrix operand order repaired and checked by reading, not compiled, not run on hardware
//
#include <hip/hip_runtime.h>

typedef _Float16 half_t;
typedef __attribute__((ext_vector_type(16))) _Float16 v16h;
typedef __attribute__((ext_vector_type(8)))  _Float16 v8h;
typedef __attribute__((ext_vector_type(8)))  float    v8f;

#define DIM    1024
#define NHEADS 16
#define HDIM   64

__device__ inline v8f wmma_f16(v16h a, v16h b, v8f c) {
  // D = A(16x32 f16) * B(32x16 f16) + C(16x16 f32)
  return __builtin_amdgcn_wmma_f32_16x16x32_f16(
      /*neg_a=*/false, a, /*neg_b=*/false, b,
      /*c_mod=*/(short)0, c, /*reuse_a=*/false, /*reuse_b=*/false);
}

// Async global->LDS copy of 32 contiguous bytes (2 x b128). The instruction
// offset is added to BOTH the LDS and global addresses (ISA 08_async_tensor).
__device__ inline void async_copy_32B(void* lds_dst, const void* gsrc) {
  unsigned int dst = (unsigned int)(unsigned long long)lds_dst;  // LDS offset = addr[31:0]
  unsigned long long src = (unsigned long long)gsrc;
  asm volatile(
      "global_load_async_to_lds_b128 %0, %1, off\n\t"
      "global_load_async_to_lds_b128 %0, %1, off offset:16"
      :: "v"(dst), "v"(src) : "memory");
}

__device__ inline void wait_async_all() {
  asm volatile("s_wait_asynccnt 0x0" ::: "memory");
}

// ---------------------------------------------------------------------------
// GEMM: out[M x N] = X[M x K] @ W^T + bias,  W is [N x K] row-major fp32.
// XT = float (convert to f16) or _Float16 (async-copy direct).
// HEAD_LAYOUT: scatter output element (gm, gn) to [(b,h), s, dh] f16 layout.
// Block: 256 thr = 8 waves. Tile: 128(M) x 64(N), BK=32, double-buffered LDS.
// ---------------------------------------------------------------------------
template <typename XT, typename OutT, bool HEAD_LAYOUT>
__global__ void __launch_bounds__(256)
gemm_bias_wmma(const XT* __restrict__ X, const float* __restrict__ W,
               const float* __restrict__ bias, OutT* __restrict__ out,
               int M, int N, int K, int seq) {
  __shared__ half_t Ash[2][128][40];  // [buf][m][k], stride 40 halfs (16B aligned)
  __shared__ half_t Bt[2][64][40];    // [buf][n][k]

  const int tid  = threadIdx.x;
  const int lane = tid & 31;
  const int wave = tid >> 5;
  const int wm   = wave & 3;   // 0..3 -> 32-row stripe
  const int wn   = wave >> 2;  // 0..1 -> 32-col stripe
  const int m0   = blockIdx.y * 128;
  const int n0   = blockIdx.x * 64;

  const int lrow  = lane & 15;
  const int kb8   = (lane >> 4) * 8;   // unused; A-frag also reads at kb16
  const int kb16  = (lane >> 4) * 16;  // B-frag k sub-base

  const int ar = tid >> 1;             // A staging: 2 threads/row
  const int ac = (tid & 1) * 16;
  const int bn = tid >> 2;             // B staging: 4 threads/row
  const int bc = (tid & 3) * 8;

  v8f acc[2][2];
#pragma unroll
  for (int mi = 0; mi < 2; ++mi)
#pragma unroll
    for (int ni = 0; ni < 2; ++ni)
#pragma unroll
      for (int i = 0; i < 8; ++i) acc[mi][ni][i] = 0.f;

  // ---- prologue: stage k-tile 0 into buffer 0 ----
  {
    const XT* src = X + (size_t)(m0 + ar) * K + ac;
    if constexpr (sizeof(XT) == 4) {
      float t[16];
      *(float4*)&t[0]  = *(const float4*)(src + 0);
      *(float4*)&t[4]  = *(const float4*)(src + 4);
      *(float4*)&t[8]  = *(const float4*)(src + 8);
      *(float4*)&t[12] = *(const float4*)(src + 12);
      v8h h0, h1;
#pragma unroll
      for (int i = 0; i < 8; ++i) { h0[i] = (half_t)t[i]; h1[i] = (half_t)t[8 + i]; }
      *(v8h*)&Ash[0][ar][ac]     = h0;
      *(v8h*)&Ash[0][ar][ac + 8] = h1;
    } else {
      async_copy_32B(&Ash[0][ar][ac], src);
    }
    const float* wsrc = W + (size_t)(n0 + bn) * K + bc;
    float t[8];
    *(float4*)&t[0] = *(const float4*)(wsrc);
    *(float4*)&t[4] = *(const float4*)(wsrc + 4);
    v8h h;
#pragma unroll
    for (int i = 0; i < 8; ++i) h[i] = (half_t)t[i];
    *(v8h*)&Bt[0][bn][bc] = h;
  }
  if constexpr (sizeof(XT) == 2) wait_async_all();
  __syncthreads();

  const int KT = K / 32;
  for (int kt = 0; kt < KT; ++kt) {
    const int cur = kt & 1;
    const int nxt = cur ^ 1;
    const bool has_next = (kt + 1) < KT;

    // ---- issue next-tile global loads first (overlap with WMMA below) ----
    float ta[16], tw[8];
    if (has_next) {
      const int k0n = (kt + 1) * 32;
      if constexpr (sizeof(XT) == 4) {
        const XT* src = X + (size_t)(m0 + ar) * K + k0n + ac;
        *(float4*)&ta[0]  = *(const float4*)(src + 0);
        *(float4*)&ta[4]  = *(const float4*)(src + 4);
        *(float4*)&ta[8]  = *(const float4*)(src + 8);
        *(float4*)&ta[12] = *(const float4*)(src + 12);
        if (k0n + 32 < K) __builtin_prefetch(src + 32, 0, 0);  // global_prefetch_b8
      } else {
        const XT* src = X + (size_t)(m0 + ar) * K + k0n + ac;
        async_copy_32B(&Ash[nxt][ar][ac], src);
      }
      const float* wsrc = W + (size_t)(n0 + bn) * K + k0n + bc;
      *(float4*)&tw[0] = *(const float4*)(wsrc);
      *(float4*)&tw[4] = *(const float4*)(wsrc + 4);
      if (k0n + 32 < K) __builtin_prefetch(wsrc + 32, 0, 0);
    }

    // ---- fragments + WMMA on current buffer ----
    v16h afr[2], bfr[2];
#pragma unroll
    for (int mi = 0; mi < 2; ++mi) {
      const half_t* ap = &Ash[cur][32 * wm + 16 * mi + lrow][0];
      v8h lo = *(const v8h*)(ap + kb16);
      v8h hi = *(const v8h*)(ap + kb16 + 8);
#pragma unroll
      for (int i = 0; i < 8; ++i) { afr[mi][i] = lo[i]; afr[mi][8 + i] = hi[i]; }
    }
#pragma unroll
    for (int ni = 0; ni < 2; ++ni) {
      const half_t* bp = &Bt[cur][32 * wn + 16 * ni + lrow][0];
      v8h lo = *(const v8h*)(bp + kb16);
      v8h hi = *(const v8h*)(bp + kb16 + 8);
#pragma unroll
      for (int i = 0; i < 8; ++i) { bfr[ni][i] = lo[i]; bfr[ni][8 + i] = hi[i]; }
    }
#pragma unroll
    for (int mi = 0; mi < 2; ++mi)
#pragma unroll
      for (int ni = 0; ni < 2; ++ni)
        acc[mi][ni] = wmma_f16(afr[mi], bfr[ni], acc[mi][ni]);

    // ---- convert + store the next tile into the other buffer ----
    if (has_next) {
      if constexpr (sizeof(XT) == 4) {
        v8h h0, h1;
#pragma unroll
        for (int i = 0; i < 8; ++i) { h0[i] = (half_t)ta[i]; h1[i] = (half_t)ta[8 + i]; }
        *(v8h*)&Ash[nxt][ar][ac]     = h0;
        *(v8h*)&Ash[nxt][ar][ac + 8] = h1;
      }
      v8h h;
#pragma unroll
      for (int i = 0; i < 8; ++i) h[i] = (half_t)tw[i];
      *(v8h*)&Bt[nxt][bn][bc] = h;
    }
    if constexpr (sizeof(XT) == 2) wait_async_all();
    __syncthreads();
  }

  // ---- epilogue: bias + store ----
  const int mb = (lane >> 4) * 8;
#pragma unroll
  for (int mi = 0; mi < 2; ++mi) {
#pragma unroll
    for (int ni = 0; ni < 2; ++ni) {
      const int gn = n0 + 32 * wn + 16 * ni + lrow;
      const float bv = bias[gn];
#pragma unroll
      for (int i = 0; i < 8; ++i) {
        const int gm = m0 + 32 * wm + 16 * mi + mb + i;
        const float v = acc[mi][ni][i] + bv;
        if constexpr (HEAD_LAYOUT) {
          const int bb = gm / seq;
          const int ss = gm - bb * seq;
          const int hh = gn >> 6;
          const int dd = gn & 63;
          out[(((size_t)(bb * NHEADS + hh) * seq) + ss) * HDIM + dd] = (OutT)v;
        } else {
          out[(size_t)gm * N + gn] = (OutT)v;
        }
      }
    }
  }
}

// ---------------------------------------------------------------------------
// In-place RoPE on f16 buffer laid out [B*H][seq][64]. One thread per pair.
// ---------------------------------------------------------------------------
__global__ void __launch_bounds__(256)
rope_kernel(half_t* __restrict__ buf, int seq) {
  const int idx  = blockIdx.x * blockDim.x + threadIdx.x;
  const int pair = idx & 31;
  const int t    = idx >> 5;
  const int s    = t % seq;
  const int bh   = t / seq;
  const float inv = __powf(10000.f, -(float)(2 * pair) / 64.f);
  const float ang = (float)s * inv;
  float sn, cs;
  __sincosf(ang, &sn, &cs);
  half_t* p = buf + ((size_t)bh * seq + s) * HDIM + 2 * pair;
  const float x0 = (float)p[0];
  const float x1 = (float)p[1];
  p[0] = (half_t)(x0 * cs - x1 * sn);
  p[1] = (half_t)(x1 * cs + x0 * sn);
}

// ---------------------------------------------------------------------------
// Flash attention: grid (L/128, B*H), 256 threads (8 waves). Each wave owns a
// 16-row L stripe (128 rows/block); iterate S in 64-wide tiles with online
// softmax. Double-buffered LDS: K tile via async global->LDS issued one tile
// ahead; V tile global-loaded one tile ahead and register-transposed after the
// WMMAs. Q/K/V: f16 [B*H][seq][64].  O: f16 [B*L][DIM] (head-major columns).
// ---------------------------------------------------------------------------
__global__ void __launch_bounds__(256)
flash_attn_kernel(const half_t* __restrict__ Q, const half_t* __restrict__ K,
                  const half_t* __restrict__ V, const unsigned char* __restrict__ mask,
                  half_t* __restrict__ O, int L, int S) {
  __shared__ half_t Ksh[2][64][80];   // [buf][s_local][dh]
  __shared__ half_t Vt[2][64][80];    // [buf][dh][s_local]  (transposed V tile)
  __shared__ half_t Psh[8][16][80];   // per-wave P staging (C-layout -> A-layout)

  const int tid  = threadIdx.x;
  const int lane = tid & 31;
  const int w    = tid >> 5;
  const int bh   = blockIdx.y;
  const int b    = bh / NHEADS;
  const int h    = bh - b * NHEADS;
  const int l0   = blockIdx.x * 128;

  const int lrow = lane & 15;
  const int kb8  = (lane >> 4) * 8;
  const int kb16 = (lane >> 4) * 16;
  const int mb   = (lane >> 4) * 8;
  const float scale = 0.125f;  // 1/sqrt(64)

  const int sr = tid >> 2;        // staging: 64 rows, 4 threads/row
  const int scol = (tid & 3) * 16;  // 16 halfs = 32B per thread

  // ---- Q A-fragments (held for whole kernel) ----
  v16h aq[2];
  {
    const half_t* qp = Q + ((size_t)bh * L + l0 + 16 * w + lrow) * HDIM;
#pragma unroll
    for (int kc = 0; kc < 2; ++kc) {
      v8h lo = *(const v8h*)(qp + 32 * kc + kb16);
      v8h hi = *(const v8h*)(qp + 32 * kc + kb16 + 8);
#pragma unroll
      for (int i = 0; i < 8; ++i) { aq[kc][i] = lo[i]; aq[kc][8 + i] = hi[i]; }
    }
  }

  float m_i[8], l_i[8];
#pragma unroll
  for (int i = 0; i < 8; ++i) { m_i[i] = -1e30f; l_i[i] = 0.f; }
  v8f acc[4];
#pragma unroll
  for (int nt = 0; nt < 4; ++nt)
#pragma unroll
    for (int i = 0; i < 8; ++i) acc[nt][i] = 0.f;

  // ---- prologue: stage s-tile 0 into buffer 0 ----
  {
    const half_t* kp = K + ((size_t)bh * S + sr) * HDIM + scol;
    async_copy_32B(&Ksh[0][sr][scol], kp);
    const half_t* vp = V + ((size_t)bh * S + sr) * HDIM + scol;
    v8h t0 = *(const v8h*)(vp);
    v8h t1 = *(const v8h*)(vp + 8);
#pragma unroll
    for (int j = 0; j < 8; ++j) {
      Vt[0][scol + j][sr]     = t0[j];
      Vt[0][scol + 8 + j][sr] = t1[j];
    }
  }
  wait_async_all();
  __syncthreads();

  const int T = S / 64;
  for (int t = 0; t < T; ++t) {
    const int cur = t & 1;
    const int nxt = cur ^ 1;
    const int s0  = t * 64;
    const bool has_next = (t + 1) < T;

    // ---- issue next tile's loads first (overlap with WMMA below) ----
    v8h nv0 = {}, nv1 = {};
    if (has_next) {
      const half_t* kp = K + ((size_t)bh * S + s0 + 64 + sr) * HDIM + scol;
      async_copy_32B(&Ksh[nxt][sr][scol], kp);
      const half_t* vp = V + ((size_t)bh * S + s0 + 64 + sr) * HDIM + scol;
      nv0 = *(const v8h*)(vp);
      nv1 = *(const v8h*)(vp + 8);
    }

    // ---- key padding mask (bool bytes) ----
    float mneg[4];
#pragma unroll
    for (int nt = 0; nt < 4; ++nt) {
      unsigned char mv = mask[(size_t)b * S + s0 + 16 * nt + lrow];
      mneg[nt] = mv ? 0.f : -1e30f;
    }

    // ---- scores: S_tile = Q (16x64) @ K_tile^T (64x64) ----
    v8f sc[4];
#pragma unroll
    for (int nt = 0; nt < 4; ++nt)
#pragma unroll
      for (int i = 0; i < 8; ++i) sc[nt][i] = 0.f;
#pragma unroll
    for (int nt = 0; nt < 4; ++nt) {
#pragma unroll
      for (int kc = 0; kc < 2; ++kc) {
        const half_t* bp = &Ksh[cur][16 * nt + lrow][32 * kc + kb16];
        v8h lo = *(const v8h*)(bp);
        v8h hi = *(const v8h*)(bp + 8);
        v16h bk;
#pragma unroll
        for (int i = 0; i < 8; ++i) { bk[i] = lo[i]; bk[8 + i] = hi[i]; }
        sc[nt] = wmma_f16(aq[kc], bk, sc[nt]);
      }
    }

    // ---- online softmax (rows striped: lane group holds 8 rows) ----
    float p[4][8], rmax[8];
#pragma unroll
    for (int i = 0; i < 8; ++i) rmax[i] = -1e30f;
#pragma unroll
    for (int nt = 0; nt < 4; ++nt)
#pragma unroll
      for (int i = 0; i < 8; ++i) {
        float sv = sc[nt][i] * scale + mneg[nt];
        p[nt][i] = sv;
        rmax[i] = fmaxf(rmax[i], sv);
      }
#pragma unroll
    for (int off = 8; off >= 1; off >>= 1)
#pragma unroll
      for (int i = 0; i < 8; ++i)
        rmax[i] = fmaxf(rmax[i], __shfl_xor(rmax[i], off, 32));

    float corr[8], rsum[8];
#pragma unroll
    for (int i = 0; i < 8; ++i) {
      float mn = fmaxf(m_i[i], rmax[i]);
      corr[i] = __expf(m_i[i] - mn);
      m_i[i] = mn;
      rsum[i] = 0.f;
    }
#pragma unroll
    for (int nt = 0; nt < 4; ++nt)
#pragma unroll
      for (int i = 0; i < 8; ++i) {
        p[nt][i] = __expf(p[nt][i] - m_i[i]);
        rsum[i] += p[nt][i];
      }
#pragma unroll
    for (int off = 8; off >= 1; off >>= 1)
#pragma unroll
      for (int i = 0; i < 8; ++i) rsum[i] += __shfl_xor(rsum[i], off, 32);
#pragma unroll
    for (int i = 0; i < 8; ++i) l_i[i] = l_i[i] * corr[i] + rsum[i];
#pragma unroll
    for (int nt = 0; nt < 4; ++nt)
#pragma unroll
      for (int i = 0; i < 8; ++i) acc[nt][i] *= corr[i];

    // ---- stage P (C-layout regs -> per-wave LDS, re-read as A-fragments) ----
#pragma unroll
    for (int nt = 0; nt < 4; ++nt)
#pragma unroll
      for (int i = 0; i < 8; ++i)
        Psh[w][mb + i][16 * nt + lrow] = (half_t)p[nt][i];

    v16h ap[2];
#pragma unroll
    for (int kc = 0; kc < 2; ++kc) {
      const half_t* pp = &Psh[w][lrow][0];
      v8h lo = *(const v8h*)(pp + 32 * kc + kb16);
      v8h hi = *(const v8h*)(pp + 32 * kc + kb16 + 8);
#pragma unroll
      for (int i = 0; i < 8; ++i) { ap[kc][i] = lo[i]; ap[kc][8 + i] = hi[i]; }
    }

    // ---- O_tile += P (16x64) @ V_tile (64x64) ----
#pragma unroll
    for (int nt = 0; nt < 4; ++nt) {
#pragma unroll
      for (int kc = 0; kc < 2; ++kc) {
        const half_t* vb = &Vt[cur][16 * nt + lrow][32 * kc + kb16];
        v8h lo = *(const v8h*)(vb);
        v8h hi = *(const v8h*)(vb + 8);
        v16h bv;
#pragma unroll
        for (int i = 0; i < 8; ++i) { bv[i] = lo[i]; bv[8 + i] = hi[i]; }
        acc[nt] = wmma_f16(ap[kc], bv, acc[nt]);
      }
    }

    // ---- register-transpose next V tile into the other buffer ----
    if (has_next) {
#pragma unroll
      for (int j = 0; j < 8; ++j) {
        Vt[nxt][scol + j][sr]     = nv0[j];
        Vt[nxt][scol + 8 + j][sr] = nv1[j];
      }
    }
    wait_async_all();
    __syncthreads();
  }

  // ---- epilogue: normalize and store to [B*L][DIM] (head-major columns) ----
#pragma unroll
  for (int nt = 0; nt < 4; ++nt) {
#pragma unroll
    for (int i = 0; i < 8; ++i) {
      const int row = l0 + 16 * w + mb + i;
      const int col = h * HDIM + 16 * nt + lrow;
      const float o = acc[nt][i] / l_i[i];
      O[((size_t)b * L + row) * DIM + col] = (half_t)o;
    }
  }
}

// ---------------------------------------------------------------------------
extern "C" void kernel_launch(void* const* d_in, const int* in_sizes, int n_in,
                              void* d_out, int out_size, void* d_ws, size_t ws_size,
                              hipStream_t stream) {
  (void)in_sizes; (void)n_in; (void)out_size; (void)ws_size;
  const float* x            = (const float*)d_in[0];
  const float* enc          = (const float*)d_in[1];
  const unsigned char* mask = (const unsigned char*)d_in[2];
  const float* Wq = (const float*)d_in[3];
  const float* bq = (const float*)d_in[4];
  const float* Wk = (const float*)d_in[5];
  const float* bk = (const float*)d_in[6];
  const float* Wv = (const float*)d_in[7];
  const float* bv = (const float*)d_in[8];
  const float* Wo = (const float*)d_in[9];
  const float* bo = (const float*)d_in[10];
  float* out = (float*)d_out;

  const int B = 2, L = 2048, S = 2048;
  const int M = B * L;  // 4096 rows for every GEMM here

  half_t* qh   = (half_t*)d_ws;                          // [B,H,L,Dh] f16
  half_t* kh   = qh + (size_t)B * NHEADS * S * HDIM;     // [B,H,S,Dh] f16
  half_t* vh   = kh + (size_t)B * NHEADS * S * HDIM;     // [B,H,S,Dh] f16
  half_t* attn = vh + (size_t)B * NHEADS * S * HDIM;     // [B*L, DIM] f16

  dim3 gblk(256);
  dim3 ggrid(DIM / 64, M / 128);
  // Q/K/V projections (bias fused, write f16 per-head layout)
  gemm_bias_wmma<float, half_t, true><<<ggrid, gblk, 0, stream>>>(x,   Wq, bq, qh, M, DIM, DIM, L);
  gemm_bias_wmma<float, half_t, true><<<ggrid, gblk, 0, stream>>>(enc, Wk, bk, kh, M, DIM, DIM, S);
  gemm_bias_wmma<float, half_t, true><<<ggrid, gblk, 0, stream>>>(enc, Wv, bv, vh, M, DIM, DIM, S);

  // RoPE on Q and K (in place)
  rope_kernel<<<(B * NHEADS * L * 32) / 256, 256, 0, stream>>>(qh, L);
  rope_kernel<<<(B * NHEADS * S * 32) / 256, 256, 0, stream>>>(kh, S);

  // Flash attention
  dim3 agrid(L / 128, B * NHEADS);
  flash_attn_kernel<<<agrid, dim3(256), 0, stream>>>(qh, kh, vh, mask, attn, L, S);

  // Output projection -> f32 d_out
  gemm_bias_wmma<half_t, float, false><<<ggrid, gblk, 0, stream>>>(attn, Wo, bo, out, M, DIM, DIM, 0);
}
